// StereoGated_53996328845816
// MI455X (gfx1250) — matrix operand order repaired and checked by reading, not compiled, not run on hardware
//
#include <hip/hip_runtime.h>
#include <math.h>

// CDNA5 (gfx1250) wave32 WMMA types
typedef __attribute__((ext_vector_type(16))) __bf16        v16bf;
typedef __attribute__((ext_vector_type(8)))  float         v8f;
typedef __attribute__((ext_vector_type(4)))  unsigned int  u32x4;
typedef __attribute__((ext_vector_type(8)))  int           i32x8;
typedef __attribute__((ext_vector_type(4)))  int           i32x4;

static constexpr int Bsz = 128;
static constexpr int Ssz = 1024;
static constexpr int Hsz = 512;   // == I

#if __has_builtin(__builtin_amdgcn_tensor_load_to_lds)
#define HAVE_TDM 1
#else
#define HAVE_TDM 0
#endif

// float -> bf16 round-to-nearest-even
__device__ __forceinline__ unsigned short f2bf(float f) {
    unsigned u = __builtin_bit_cast(unsigned, f);
    unsigned r = u + 0x7FFFu + ((u >> 16) & 1u);
    return (unsigned short)(r >> 16);
}

#if HAVE_TDM
// ---------------------------------------------------------------------------
// Issue a 1-D TDM copy of `bytes` (multiple of 8) from global `gptr` into LDS
// at byte offset `lds_off`. D# layout per CDNA5 ISA:
//   group0: count=1 | lds_addr | global_addr[56:0] | type=2
//   group1: data_size=3 (8B), tensor_dim0 = tile_dim0 = stride = n elements
// Wave-uniform arguments; tracked by TENSORcnt.
// ---------------------------------------------------------------------------
__device__ __forceinline__ void tdm_load_1d(unsigned lds_off, const void* gptr,
                                            unsigned bytes) {
    unsigned n = bytes >> 3;                       // 8-byte elements
    unsigned long long ga = (unsigned long long)gptr;
    u32x4 g0;
    g0.x = 1u;                                     // count=1 (valid user D#)
    g0.y = lds_off;                                // lds_addr
    g0.z = (unsigned)ga;                           // global_addr[31:0]
    g0.w = (unsigned)(ga >> 32) | (2u << 30);      // global_addr[56:32] | type=2
    i32x8 g1;
    g1[0] = (int)(3u << 16);                       // data_size=3 (8 bytes)
    g1[1] = (int)(n << 16);                        // tensor_dim0[15:0] (bits 79:64... packed low)
    g1[2] = 0;                                     // tensor_dim0 hi / tensor_dim1
    g1[3] = (int)(n << 16);                        // tile_dim0
    g1[4] = 0;                                     // tile_dim1/tile_dim2 unused
    g1[5] = (int)n;                                // tensor_dim0_stride[31:0]
    g1[6] = 0;
    g1[7] = 0;
    i32x4 gz = {0, 0, 0, 0};
#if defined(__clang_major__) && __clang_major__ >= 23
    i32x8 z8 = {0, 0, 0, 0, 0, 0, 0, 0};
    __builtin_amdgcn_tensor_load_to_lds(g0, g1, gz, gz, z8, 0);
#else
    __builtin_amdgcn_tensor_load_to_lds(g0, g1, gz, gz, 0);
#endif
}

__device__ __forceinline__ void tdm_wait0() {
#if __has_builtin(__builtin_amdgcn_s_wait_tensorcnt)
    __builtin_amdgcn_s_wait_tensorcnt(0);
#else
    asm volatile("s_wait_tensorcnt 0x0" ::: "memory");
#endif
}
#endif  // HAVE_TDM

// -----------------------------------------------------------------------------
// Pack a 512x512 row-major f32 weight (K x N) into the CDNA5 bf16 B-fragment
// layout: [kt(16)][nt(32)][lane(32)][j(16)] ushort.
//   lane <  16 : N = lane,      K = kt*32 + (j<8 ? j : j+8)      (j = 0..15)
//   lane >= 16 : N = lane - 16, K = kt*32 + 8 + (j<8 ? j : j+8)
// -----------------------------------------------------------------------------
__global__ __launch_bounds__(256) void pack_b_bf16(const float* __restrict__ W,
                                                   unsigned short* __restrict__ P) {
    int e = blockIdx.x * 256 + threadIdx.x;            // 0 .. 512*512-1
    int j    =  e        & 15;
    int lane = (e >> 4)  & 31;
    int nt   = (e >> 9)  & 31;
    int kt   =  e >> 14;
    int k = kt * 32 + ((j < 8) ? j : j + 8) + ((lane >= 16) ? 8 : 0);
    int n = nt * 16 + (lane & 15);
    P[e] = f2bf(W[k * Hsz + n]);
}

// -----------------------------------------------------------------------------
// Thin projections: xr = X @ W_ir + br, xi = X @ W_ii + bi. One wave per row.
// -----------------------------------------------------------------------------
__global__ __launch_bounds__(256) void proj_ri(const float* __restrict__ X,
                                               const float* __restrict__ Wr,
                                               const float* __restrict__ Wi,
                                               const float* __restrict__ br,
                                               const float* __restrict__ bi,
                                               float* __restrict__ xr,
                                               float* __restrict__ xi) {
    int row  = blockIdx.x * 8 + (threadIdx.x >> 5);
    int lane = threadIdx.x & 31;
    const float* x = X + (size_t)row * Hsz;
    float s1 = 0.f, s2 = 0.f;
    for (int k = lane; k < Hsz; k += 32) {
        float v = x[k];
        s1 += v * Wr[k];
        s2 += v * Wi[k];
    }
    #pragma unroll
    for (int o = 16; o >= 1; o >>= 1) {
        s1 += __shfl_xor(s1, o, 32);
        s2 += __shfl_xor(s2, o, 32);
    }
    if (lane == 0) { xr[row] = s1 + br[0]; xi[row] = s2 + bi[0]; }
}

// -----------------------------------------------------------------------------
// Phase-1 GEMM: xh = inputs @ W_ih + bias_h -> staged into d_out.
// WG = 256 thr (8 waves), tile 128 rows x 128 cols. All 8 waves share the same
// 8 B-tiles per k-step, so B is staged through LDS by the Tensor Data Mover
// (double-buffered 8KB slices, one wave issues the DMA, TENSORcnt + barrier
// for completion). A fragments converted f32->bf16 on the fly from global.
// -----------------------------------------------------------------------------
__global__ __launch_bounds__(256) void gemm_xh(const float* __restrict__ X,
                                               const unsigned short* __restrict__ Wp,
                                               const float* __restrict__ bias,
                                               float* __restrict__ Y) {
    __shared__ __align__(16) unsigned short Bst[2][8][32][16];   // 16 KB staging

    const int lane  = threadIdx.x & 31;
    const int wave  = threadIdx.x >> 5;
    const int rBase = blockIdx.x * 128 + wave * 16;
    const int cBase = blockIdx.y * 128;
    const int cb    = cBase >> 4;                 // first n-tile index
    const int m     = lane & 15;
    const int off   = (lane >= 16) ? 8 : 0;       // A-fragment K half select

    v8f acc[8] = {};
    const float* xrow = X + (size_t)(rBase + m) * Hsz;

#if HAVE_TDM
    // prime buffer 0 with kt = 0 (8 n-tiles * 1KB, contiguous in packed layout)
    if (threadIdx.x < 32) {
        tdm_load_1d((unsigned)(unsigned long long)&Bst[0][0][0][0],
                    Wp + ((size_t)cb << 9), 8192);
        tdm_wait0();
    }
    __syncthreads();
#endif

    for (int kt = 0; kt < 16; ++kt) {
        const int cur = kt & 1;
#if HAVE_TDM
        if (kt < 15 && threadIdx.x < 32) {
            tdm_load_1d((unsigned)(unsigned long long)&Bst[cur ^ 1][0][0][0],
                        Wp + (((size_t)(kt + 1) * 32 + cb) << 9), 8192);
        }
#endif
        // A fragment (16x32 bf16): lane half -> K in {off..off+7, 16+off..23+off}
        int k0 = kt * 32 + off;
        float4 f0 = *(const float4*)(xrow + k0);
        float4 f1 = *(const float4*)(xrow + k0 + 4);
        float4 f2 = *(const float4*)(xrow + k0 + 16);
        float4 f3 = *(const float4*)(xrow + k0 + 20);
        union { unsigned short s[16]; v16bf v; } a;
        a.s[0]  = f2bf(f0.x); a.s[1]  = f2bf(f0.y); a.s[2]  = f2bf(f0.z); a.s[3]  = f2bf(f0.w);
        a.s[4]  = f2bf(f1.x); a.s[5]  = f2bf(f1.y); a.s[6]  = f2bf(f1.z); a.s[7]  = f2bf(f1.w);
        a.s[8]  = f2bf(f2.x); a.s[9]  = f2bf(f2.y); a.s[10] = f2bf(f2.z); a.s[11] = f2bf(f2.w);
        a.s[12] = f2bf(f3.x); a.s[13] = f2bf(f3.y); a.s[14] = f2bf(f3.z); a.s[15] = f2bf(f3.w);

        #pragma unroll
        for (int nt = 0; nt < 8; ++nt) {
#if HAVE_TDM
            v16bf b = *reinterpret_cast<const v16bf*>(&Bst[cur][nt][lane][0]);
#else
            v16bf b = *reinterpret_cast<const v16bf*>(
                Wp + ((((size_t)kt * 32 + cb + nt) * 32 + lane) << 4));
#endif
            acc[nt] = __builtin_amdgcn_wmma_f32_16x16x32_bf16(
                false, a.v, false, b, (short)0, acc[nt], false, false);
        }
#if HAVE_TDM
        if (threadIdx.x < 32) tdm_wait0();   // next slice landed in LDS
        __syncthreads();                     // everyone done reading cur; next ready
#endif
    }

    const int mo = (lane >= 16) ? 8 : 0;
    #pragma unroll
    for (int nt = 0; nt < 8; ++nt) {
        int col = cBase + nt * 16 + (lane & 15);
        float bv = bias[col];
        #pragma unroll
        for (int i = 0; i < 8; ++i) {
            int row = rBase + mo + i;
            Y[(size_t)row * Hsz + col] = acc[nt][i] + bv;
        }
    }
}

// -----------------------------------------------------------------------------
// Persistent recurrence. 8 workgroups x 256 threads; WG owns 16 batch rows,
// h kept in LDS for all 1024 steps. Waves use disjoint W_hh columns (no
// intra-WG reuse), so B streams L2->registers, software-pipelined (double
// buffered) so fragment loads for kt+1 overlap the WMMAs of kt.
// -----------------------------------------------------------------------------
__global__ __launch_bounds__(256) void recurrent(const unsigned short* __restrict__ Whp,
                                                 const float* __restrict__ Whr,
                                                 const float* __restrict__ Whi,
                                                 const float* __restrict__ xr,
                                                 const float* __restrict__ xi,
                                                 float* __restrict__ out) {
    __shared__ float h[16][Hsz];                               // 32 KB fp32 state
    __shared__ __align__(32) unsigned short rh[16][32][16];    // 16 KB A fragments
    __shared__ float red[2][16][16];
    __shared__ float zArr[16], rArr[16];

    const int tid   = threadIdx.x;
    const int lane  = tid & 31;
    const int wave  = tid >> 5;
    const int bBase = blockIdx.x * 16;

    for (int e = tid; e < 16 * Hsz; e += 256) h[e >> 9][e & 511] = 0.f;
    __syncthreads();

    for (int t = 0; t < Ssz; ++t) {
        // ---- scalar gates: z = sig(xr + h.Whr), r = sig(xi + h.Whi) ----
        {
            int b = tid >> 4, part = tid & 15;
            float s1 = 0.f, s2 = 0.f;
            const float* hb = h[b];
            int k0 = part * 32;
            #pragma unroll 8
            for (int i = 0; i < 32; ++i) {
                float v = hb[k0 + i];
                s1 += v * Whr[k0 + i];
                s2 += v * Whi[k0 + i];
            }
            red[0][b][part] = s1;
            red[1][b][part] = s2;
        }
        __syncthreads();
        if (tid < 16) {
            int b = tid;
            float s1 = 0.f, s2 = 0.f;
            #pragma unroll
            for (int p = 0; p < 16; ++p) { s1 += red[0][b][p]; s2 += red[1][b][p]; }
            size_t idx = (size_t)(bBase + b) * Ssz + t;
            zArr[b] = 1.f / (1.f + __expf(-(xr[idx] + s1)));
            rArr[b] = 1.f / (1.f + __expf(-(xi[idx] + s2)));
        }
        __syncthreads();

        // ---- pack A = r*h into bf16 16x32 fragments (16 k-tiles) ----
        for (int e = tid; e < 8192; e += 256) {
            int j = e & 15, ln = (e >> 4) & 31, kt = e >> 9;
            int mm = ln & 15;
            int kl = ((j < 8) ? j : j + 8) + ((ln >= 16) ? 8 : 0);
            rh[kt][ln][j] = f2bf(rArr[mm] * h[mm][kt * 32 + kl]);
        }
        __syncthreads();

        // ---- WMMA, wave owns cols [wave*64, +64); pipelined B stream ----
        v8f acc[4] = {};
        v16bf bcur[4], bnxt[4];
        #pragma unroll
        for (int nt = 0; nt < 4; ++nt)
            bcur[nt] = *reinterpret_cast<const v16bf*>(
                Whp + ((((size_t)(wave * 4 + nt)) * 32 + lane) << 4));
        #pragma unroll
        for (int kt = 0; kt < 16; ++kt) {
            if (kt < 15) {
                #pragma unroll
                for (int nt = 0; nt < 4; ++nt)
                    bnxt[nt] = *reinterpret_cast<const v16bf*>(
                        Whp + ((((size_t)(kt + 1) * 32 + wave * 4 + nt) * 32 + lane) << 4));
            }
            v16bf a = *reinterpret_cast<const v16bf*>(&rh[kt][lane][0]);
            #pragma unroll
            for (int nt = 0; nt < 4; ++nt)
                acc[nt] = __builtin_amdgcn_wmma_f32_16x16x32_bf16(
                    false, a, false, bcur[nt], (short)0, acc[nt], false, false);
            #pragma unroll
            for (int nt = 0; nt < 4; ++nt) bcur[nt] = bnxt[nt];
        }

        // ---- epilogue: n = tanh(xh + acc); h = z*h + (1-z)*n ----
        const int mo = (lane >= 16) ? 8 : 0;
        #pragma unroll
        for (int nt = 0; nt < 4; ++nt) {
            int col = wave * 64 + nt * 16 + (lane & 15);
            #pragma unroll
            for (int i = 0; i < 8; ++i) {
                int mm = mo + i;
                int bG = bBase + mm;
                size_t idx = ((size_t)bG * Ssz + t) * Hsz + col;
                float n  = tanhf(out[idx] + acc[nt][i]);   // out[idx] holds xh
                float z  = zArr[mm];
                float hv = z * h[mm][col] + (1.f - z) * n; // (mm,col) exclusive
                out[idx] = hv;
                h[mm][col] = hv;
                if (t == Ssz - 1)
                    out[(size_t)Bsz * Ssz * Hsz + (size_t)bG * Hsz + col] = hv;
            }
        }
        __syncthreads();   // h fully updated before next step's gate dots
    }
}

// -----------------------------------------------------------------------------
extern "C" void kernel_launch(void* const* d_in, const int* in_sizes, int n_in,
                              void* d_out, int out_size, void* d_ws, size_t ws_size,
                              hipStream_t stream) {
    const float* inputs = (const float*)d_in[0];
    const float* W_ih   = (const float*)d_in[1];
    const float* W_hh   = (const float*)d_in[2];
    const float* W_ir   = (const float*)d_in[3];
    const float* W_hr   = (const float*)d_in[4];
    const float* W_ii   = (const float*)d_in[5];
    const float* W_hi   = (const float*)d_in[6];
    const float* bias_h = (const float*)d_in[7];
    const float* bias_r = (const float*)d_in[8];
    const float* bias_i = (const float*)d_in[9];
    float* out = (float*)d_out;

    // workspace layout (~2 MB total)
    unsigned short* WihP = (unsigned short*)d_ws;                       // 512 KB
    unsigned short* WhhP = WihP + 512 * 512;                            // 512 KB
    float* xr = (float*)((char*)d_ws + 2u * 512 * 512 * 2);             // 512 KB
    float* xi = xr + Bsz * Ssz;                                         // 512 KB

    // pack weights to bf16 WMMA B-fragment layout
    pack_b_bf16<<<1024, 256, 0, stream>>>(W_ih, WihP);
    pack_b_bf16<<<1024, 256, 0, stream>>>(W_hh, WhhP);

    // thin gate projections (one wave per (b,s) row)
    proj_ri<<<(Bsz * Ssz) / 8, 256, 0, stream>>>(inputs, W_ir, W_ii, bias_r, bias_i, xr, xi);

    // xh = inputs @ W_ih + bias_h, staged in d_out (TDM-fed WMMA)
    gemm_xh<<<dim3((Bsz * Ssz) / 128, Hsz / 128), 256, 0, stream>>>(inputs, WihP, bias_h, out);

    // persistent sequential scan (8 WGs, 16 batch rows each)
    recurrent<<<Bsz / 16, 256, 0, stream>>>(WhhP, W_hr, W_hi, xr, xi, out);
}
